// PrunedBlock_23055384445085
// MI455X (gfx1250) — compile-verified, matrix operand order checked by reading, not run on hardware
//
#include <hip/hip_runtime.h>
#include <stdint.h>

// ---------- types ----------
typedef __attribute__((ext_vector_type(16))) __bf16 v16bf;
typedef __attribute__((ext_vector_type(8)))  float  v8f;
typedef __attribute__((ext_vector_type(4)))  unsigned int v4u;
typedef __attribute__((ext_vector_type(8)))  int v8i;
typedef __attribute__((ext_vector_type(4)))  int v4i;

union FragAB { v16bf v; uint32_t u[8]; };

#if defined(__gfx1250__) && __has_builtin(__builtin_amdgcn_tensor_load_to_lds)
#define USE_TDM 1
#endif

__device__ __forceinline__ uint16_t f2bf(float x) {
  uint32_t u = __float_as_uint(x);
  uint32_t r = u + 0x7FFFu + ((u >> 16) & 1u);
  return (uint16_t)(r >> 16);
}
__device__ __forceinline__ float bf2f(uint16_t h) {
  return __uint_as_float(((uint32_t)h) << 16);
}

// ---------- elementwise f32 -> bf16 ----------
__global__ __launch_bounds__(256) void cvt_f32_bf16(const float* __restrict__ in,
                                                    uint16_t* __restrict__ out, int n) {
  int i = blockIdx.x * 256 + threadIdx.x;
  if (i < n) out[i] = f2bf(in[i]);
}

// ---------- LayerNorm over C=768, one row per block, bf16 out ----------
__global__ __launch_bounds__(256) void ln768_kernel(const float* __restrict__ in,
                                                    const float* __restrict__ w,
                                                    const float* __restrict__ b,
                                                    uint16_t* __restrict__ out) {
  const int C = 768;
  int row = blockIdx.x, t = threadIdx.x;
  const float* p = in + (size_t)row * C;
  float a0 = p[t], a1 = p[t + 256], a2 = p[t + 512];
  __shared__ float r1[256], r2[256];
  r1[t] = a0 + a1 + a2;
  r2[t] = a0 * a0 + a1 * a1 + a2 * a2;
  __syncthreads();
  for (int off = 128; off > 0; off >>= 1) {
    if (t < off) { r1[t] += r1[t + off]; r2[t] += r2[t + off]; }
    __syncthreads();
  }
  float mean = r1[0] * (1.0f / 768.0f);
  float var  = r2[0] * (1.0f / 768.0f) - mean * mean;
  float rstd = rsqrtf(var + 1e-6f);
  uint16_t* q = out + (size_t)row * C;
  q[t]       = f2bf((a0 - mean) * rstd * w[t]       + b[t]);
  q[t + 256] = f2bf((a1 - mean) * rstd * w[t + 256] + b[t + 256]);
  q[t + 512] = f2bf((a2 - mean) * rstd * w[t + 512] + b[t + 512]);
}

// ---------- bf16 WMMA GEMM: C[M,N] = A[M,K] @ B[K,N] + bias (+epilogue) ----------
// mode 0: f32 out = acc + bias
// mode 1: f32 out = acc + bias + res
// mode 2: bf16 out = gelu(acc + bias)
// mode 3: bf16 out = acc + bias
#define BM 128
#define BN 128
#define BK 32

__global__ __launch_bounds__(256) void gemm_wmma_bf16(
    const uint16_t* __restrict__ A, int lda,
    const uint16_t* __restrict__ Bw, int ldb,
    float* __restrict__ Cf, uint16_t* __restrict__ Cb, int ldc,
    const float* __restrict__ bias,
    const float* __restrict__ res, int ldr,
    int M, int N, int K, int mode) {
  __shared__ uint16_t sA[BM][BK + 4];   // 36 u16 = 72B row stride (16 DW + 2 DW pad)
  __shared__ uint16_t sB[BN][BK + 4];
  int tid = threadIdx.x;
  int lane = tid & 31, wid = tid >> 5;
  int wm = wid & 3, wn = wid >> 2;             // 4 x 2 wave grid
  int m0 = blockIdx.y * BM, n0 = blockIdx.x * BN;
  int mbase = wm * 32, nbase = wn * 64;        // wave tile: 32 (M) x 64 (N)
  int mrow = lane & 15, hl = lane >> 4;        // ISA 16-bit fragment layout

  v8f acc[2][4];
  v8f zero = {0.f, 0.f, 0.f, 0.f, 0.f, 0.f, 0.f, 0.f};
#pragma unroll
  for (int i = 0; i < 2; ++i)
#pragma unroll
    for (int j = 0; j < 4; ++j) acc[i][j] = zero;

  for (int k0 = 0; k0 < K; k0 += BK) {
#ifdef USE_TDM
    // ---- A tile 128x32 via Tensor Data Mover (wave 0 issues; TENSORcnt) ----
    if (wid == 0) {
      uint64_t gaddr = (uint64_t)(uintptr_t)(A + (size_t)m0 * lda + k0);
      uint32_t lds_off = (uint32_t)(uintptr_t)(&sA[0][0]);
      uint32_t dim1 = (uint32_t)(M - m0);          // rows available; OOB rows -> zero fill
      uint32_t ulda = (uint32_t)lda;
      v4u g0;
      g0[0] = 1u;                                  // count=1, no gather, user desc
      g0[1] = lds_off;                             // lds_addr
      g0[2] = (uint32_t)gaddr;                     // global_addr[31:0]
      g0[3] = ((uint32_t)(gaddr >> 32) & 0x01FFFFFFu) | (2u << 30);  // addr[56:32] | type=2
      v8i g1;
      // data_size=1(2B) | pad_enable | pad_interval=3(16DW) | pad_amount=1(2DW)
      g1[0] = (int)((1u << 16) | (1u << 20) | (3u << 22) | (1u << 25));
      g1[1] = (int)((ulda & 0xFFFFu) << 16);                              // tensor_dim0[15:0]
      g1[2] = (int)(((ulda >> 16) & 0xFFFFu) | ((dim1 & 0xFFFFu) << 16)); // dim0[31:16] | dim1[15:0]
      g1[3] = (int)(((dim1 >> 16) & 0xFFFFu) | (32u << 16));              // dim1[31:16] | tile_dim0=32
      g1[4] = (int)(128u);                                                // tile_dim1=128, tile_dim2=0
      g1[5] = (int)ulda;                                                  // tensor_dim0_stride[31:0]
      g1[6] = 0;                                                          // stride[47:32] | dim1_stride lo
      g1[7] = 0;
      v4i gz = {0, 0, 0, 0};
#if __clang_major__ >= 23
      v8i gz8 = {0, 0, 0, 0, 0, 0, 0, 0};
      __builtin_amdgcn_tensor_load_to_lds(g0, g1, gz, gz, gz8, 0);
#else
      __builtin_amdgcn_tensor_load_to_lds(g0, g1, gz, gz, 0);
#endif
    }
#else
    // ---- A tile 128x32 (bf16, K contiguous), manual path ----
#pragma unroll
    for (int it = 0; it < 4; ++it) {
      int lin = it * 256 + tid;
      int row = lin >> 3, c4 = (lin & 7) << 2;
      int gm = m0 + row;
      uint2 val = make_uint2(0u, 0u);
      if (gm < M) val = *(const uint2*)(A + (size_t)gm * lda + k0 + c4);
      *(uint32_t*)&sA[row][c4]     = val.x;
      *(uint32_t*)&sA[row][c4 + 2] = val.y;
    }
#endif
    // ---- B tile 32x128 -> stored transposed sB[n][k] ----
#pragma unroll
    for (int it = 0; it < 4; ++it) {
      int lin = it * 256 + tid;
      int kk = lin >> 5, n4 = (lin & 31) << 2;
      uint2 val = *(const uint2*)(Bw + (size_t)(k0 + kk) * ldb + n0 + n4);
      sB[n4 + 0][kk] = (uint16_t)(val.x & 0xFFFFu);
      sB[n4 + 1][kk] = (uint16_t)(val.x >> 16);
      sB[n4 + 2][kk] = (uint16_t)(val.y & 0xFFFFu);
      sB[n4 + 3][kk] = (uint16_t)(val.y >> 16);
    }
    if (k0 + BK < K) {
      __builtin_prefetch(Bw + (size_t)(k0 + BK + (tid & 31)) * ldb + n0, 0, 1);
    }
#ifdef USE_TDM
    if (wid == 0) __builtin_amdgcn_s_wait_tensorcnt(0);
#endif
    __syncthreads();

    FragAB fa[2], fb[4];
#pragma unroll
    for (int mt = 0; mt < 2; ++mt) {
      int r = mbase + mt * 16 + mrow;
#pragma unroll
      for (int i = 0; i < 4; ++i) fa[mt].u[i]     = *(const uint32_t*)&sA[r][hl * 8 + 2 * i];
#pragma unroll
      for (int i = 0; i < 4; ++i) fa[mt].u[4 + i] = *(const uint32_t*)&sA[r][16 + hl * 8 + 2 * i];
    }
#pragma unroll
    for (int nt = 0; nt < 4; ++nt) {
      int c = nbase + nt * 16 + mrow;
#pragma unroll
      for (int i = 0; i < 4; ++i) fb[nt].u[i]     = *(const uint32_t*)&sB[c][hl * 8 + 2 * i];
#pragma unroll
      for (int i = 0; i < 4; ++i) fb[nt].u[4 + i] = *(const uint32_t*)&sB[c][16 + hl * 8 + 2 * i];
    }
#pragma unroll
    for (int mt = 0; mt < 2; ++mt)
#pragma unroll
      for (int nt = 0; nt < 4; ++nt)
        acc[mt][nt] = __builtin_amdgcn_wmma_f32_16x16x32_bf16(
            false, fa[mt].v, false, fb[nt].v, (short)0, acc[mt][nt], false, false);
    __syncthreads();
  }

  // epilogue
#pragma unroll
  for (int mt = 0; mt < 2; ++mt) {
#pragma unroll
    for (int nt = 0; nt < 4; ++nt) {
      int n = n0 + nbase + nt * 16 + mrow;
#pragma unroll
      for (int r = 0; r < 8; ++r) {
        int m = m0 + mbase + mt * 16 + r + 8 * hl;
        if (m < M) {
          float v = acc[mt][nt][r];
          if (bias) v += bias[n];
          if (mode == 1) v += res[(size_t)m * ldr + n];
          if (mode == 2) {
            float inner = 0.7978845608f * (v + 0.044715f * v * v * v);
            v = 0.5f * v * (1.0f + tanhf(inner));
          }
          if (mode == 0 || mode == 1) Cf[(size_t)m * ldc + n] = v;
          else                        Cb[(size_t)m * ldc + n] = f2bf(v);
        }
      }
    }
  }
}

// ---------- CLS attention scores: softmax over keys per head, mean over heads ----------
__global__ __launch_bounds__(640) void cls_scores_kernel(const float* __restrict__ qcls,
                                                         const uint16_t* __restrict__ kall,
                                                         float* __restrict__ scores) {
  const int C = 768, T = 577, H = 12, NP = 576;
  int b = blockIdx.x, t = threadIdx.x;
  __shared__ float qc[768];
  __shared__ float sc[12 * 577];
  __shared__ float hmx[12], hsm[12];
  for (int i = t; i < C; i += 640) qc[i] = qcls[(size_t)b * C + i] * 0.125f;  // scale = hd^-0.5
  __syncthreads();
  for (int m = t; m < T; m += 640) {
    const uint16_t* kr = kall + ((size_t)b * T + m) * C;
    for (int h = 0; h < H; ++h) {
      float s = 0.f;
#pragma unroll
      for (int d = 0; d < 64; ++d) s += qc[h * 64 + d] * bf2f(kr[h * 64 + d]);
      sc[h * T + m] = s;
    }
  }
  __syncthreads();
  if (t < H) {
    float mx = -3.4e38f;
    for (int m = 0; m < T; ++m) mx = fmaxf(mx, sc[t * T + m]);
    float sm = 0.f;
    for (int m = 0; m < T; ++m) sm += __expf(sc[t * T + m] - mx);
    hmx[t] = mx; hsm[t] = sm;
  }
  __syncthreads();
  for (int m = t; m < T; m += 640) {
    if (m >= 1) {
      float a = 0.f;
      for (int h = 0; h < H; ++h) a += __expf(sc[h * T + m] - hmx[h]) / hsm[h];
      scores[(size_t)b * NP + (m - 1)] = a * (1.0f / 12.0f);
    }
  }
}

// ---------- top-288 of 576, ascending; writes index/mask outputs ----------
__global__ __launch_bounds__(576) void topk_kernel(const float* __restrict__ scores,
                                                   const int* __restrict__ pidx,
                                                   int* __restrict__ kpos,
                                                   float* __restrict__ out_idx,
                                                   float* __restrict__ out_mask) {
  const int NP = 576, KP = 288;
  int b = blockIdx.x, t = threadIdx.x;
  __shared__ float v[576], rv[576];
  __shared__ int ri[576], keep[576];
  v[t] = scores[(size_t)b * NP + t];
  keep[t] = 0;
  __syncthreads();
  for (int it = 0; it < KP; ++it) {
    rv[t] = v[t]; ri[t] = t;
    __syncthreads();
    for (int off = 512; off > 0; off >>= 1) {
      if (t < off && t + off < NP) {
        float ov = rv[t + off]; int oi = ri[t + off];
        if (ov > rv[t] || (ov == rv[t] && oi < ri[t])) { rv[t] = ov; ri[t] = oi; }
      }
      __syncthreads();
    }
    if (t == 0) { keep[ri[0]] = 1; v[ri[0]] = -3.4e38f; }
    __syncthreads();
  }
  if (t == 0) {
    int cnt = 0;
    for (int m = 0; m < NP; ++m)
      if (keep[m]) {
        kpos[b * KP + cnt] = m;
        out_idx[b * KP + cnt] = (float)pidx[b * NP + m];
        ++cnt;
      }
  }
  out_mask[(size_t)b * NP + t] = 0.f;
  __syncthreads();
  if (keep[t]) out_mask[(size_t)b * NP + pidx[b * NP + t]] = 1.f;
}

// ---------- gather kept tokens (CLS + 288 patches) ----------
__global__ __launch_bounds__(256) void gather_kernel(const float* __restrict__ tokens,
                                                     const int* __restrict__ kpos,
                                                     float* __restrict__ x) {
  const int C = 768, T = 577, TK = 289, KP = 288;
  int row = blockIdx.x, t = threadIdx.x;
  int b = row / TK, tl = row % TK;
  int src = (tl == 0) ? 0 : (kpos[b * KP + tl - 1] + 1);
  const float* p = tokens + ((size_t)b * T + src) * C;
  float* q = x + (size_t)row * C;
  q[t] = p[t]; q[t + 256] = p[t + 256]; q[t + 512] = p[t + 512];
}

// ---------- flash-style attention on kept tokens (per b,h; key chunks in LDS) ----------
__global__ __launch_bounds__(128) void attn_kernel(const uint16_t* __restrict__ qkv,
                                                   uint16_t* __restrict__ ao) {
  const int TK = 289, H = 12, HD = 64, LDQ = 2304, C = 768;
  int qc = blockIdx.x, h = blockIdx.y, b = blockIdx.z;
  int t = threadIdx.x;
  int nq = qc * 128 + t;
  bool act = nq < TK;
  __shared__ float sk[64][65];
  __shared__ float sv[64][65];
  float qr[64], accv[64];
  float mrun = -3.4e38f, lrun = 0.f;
  const size_t rowb = (size_t)b * TK;
  if (act) {
    const uint16_t* qp = qkv + (rowb + nq) * LDQ + (0 * H + h) * HD;
#pragma unroll
    for (int d = 0; d < 64; ++d) qr[d] = bf2f(qp[d]) * 0.125f;
  }
#pragma unroll
  for (int d = 0; d < 64; ++d) accv[d] = 0.f;

  for (int kc = 0; kc < TK; kc += 64) {
    int len = min(64, TK - kc);
    for (int i = t; i < 64 * 64; i += 128) {
      int j = i >> 6, d = i & 63;
      if (j < len) {
        const uint16_t* kp = qkv + (rowb + kc + j) * LDQ;
        sk[j][d] = bf2f(kp[(1 * H + h) * HD + d]);
        sv[j][d] = bf2f(kp[(2 * H + h) * HD + d]);
      }
    }
    __syncthreads();
    if (act) {
      for (int j = 0; j < len; ++j) {
        float s = 0.f;
#pragma unroll
        for (int d = 0; d < 64; ++d) s += qr[d] * sk[j][d];
        float mn = fmaxf(mrun, s);
        float corr = __expf(mrun - mn);
        float p = __expf(s - mn);
        lrun = lrun * corr + p;
#pragma unroll
        for (int d = 0; d < 64; ++d) accv[d] = accv[d] * corr + p * sv[j][d];
        mrun = mn;
      }
    }
    __syncthreads();
  }
  if (act) {
    uint16_t* op = ao + (rowb + nq) * C + h * HD;
    float inv = 1.f / lrun;
#pragma unroll
    for (int d = 0; d < 64; ++d) op[d] = f2bf(accv[d] * inv);
  }
}

// ---------- host orchestration ----------
extern "C" void kernel_launch(void* const* d_in, const int* in_sizes, int n_in,
                              void* d_out, int out_size, void* d_ws, size_t ws_size,
                              hipStream_t stream) {
  const int B = 32, T = 577, C = 768, NP = 576, KP = 288, TK = 289;
  const int M1 = B * T;   // 18464 rows for scoring
  const int M2 = B * TK;  // 9248 rows for the block

  const float* tokens = (const float*)d_in[0];
  const int*   pidx   = (const int*)d_in[1];
  const float* n1w    = (const float*)d_in[2];
  const float* n1b    = (const float*)d_in[3];
  const float* qkvw   = (const float*)d_in[4];
  const float* qkvb   = (const float*)d_in[5];
  const float* projw  = (const float*)d_in[6];
  const float* projb  = (const float*)d_in[7];
  const float* n2w    = (const float*)d_in[8];
  const float* n2b    = (const float*)d_in[9];
  const float* fc1w   = (const float*)d_in[10];
  const float* fc1b   = (const float*)d_in[11];
  const float* fc2w   = (const float*)d_in[12];
  const float* fc2b   = (const float*)d_in[13];

  char* ws = (char*)d_ws;
  size_t off = 0;
  auto alloc = [&](size_t bytes) {
    size_t o = off;
    off += (bytes + 255) & ~(size_t)255;
    return o;
  };
  size_t o_wqkv  = alloc((size_t)C * 3 * C * 2);
  size_t o_wproj = alloc((size_t)C * C * 2);
  size_t o_wfc1  = alloc((size_t)C * 4 * C * 2);
  size_t o_wfc2  = alloc((size_t)4 * C * C * 2);
  size_t regA_bytes = (size_t)M1 * C * 2 * 2;   // normA + kall (phase A)
  size_t qkv_bytes  = (size_t)M2 * 3 * C * 2;   // qkv_out (phase B, aliased)
  size_t o_regA  = alloc(regA_bytes > qkv_bytes ? regA_bytes : qkv_bytes);
  size_t o_qcls  = alloc((size_t)B * C * 4);
  size_t o_scor  = alloc((size_t)B * NP * 4);
  size_t o_kpos  = alloc((size_t)B * KP * 4);
  size_t o_x     = alloc((size_t)M2 * C * 4);
  size_t o_normx = alloc((size_t)M2 * C * 2);
  size_t o_attn  = alloc((size_t)M2 * C * 2);
  size_t o_x2    = alloc((size_t)M2 * C * 4);
  size_t o_ln2   = alloc((size_t)M2 * C * 2);
  size_t o_hact  = o_x;  // aliases x+normx+attn (free by fc1 time), exactly M2*4C*2 bytes

  uint16_t* Wqkv  = (uint16_t*)(ws + o_wqkv);
  uint16_t* Wproj = (uint16_t*)(ws + o_wproj);
  uint16_t* Wfc1  = (uint16_t*)(ws + o_wfc1);
  uint16_t* Wfc2  = (uint16_t*)(ws + o_wfc2);
  uint16_t* normA = (uint16_t*)(ws + o_regA);
  uint16_t* kall  = (uint16_t*)(ws + o_regA + (size_t)M1 * C * 2);
  uint16_t* qkvo  = (uint16_t*)(ws + o_regA);
  float*    qcls  = (float*)(ws + o_qcls);
  float*    scor  = (float*)(ws + o_scor);
  int*      kpos  = (int*)(ws + o_kpos);
  float*    x     = (float*)(ws + o_x);
  uint16_t* normx = (uint16_t*)(ws + o_normx);
  uint16_t* attn  = (uint16_t*)(ws + o_attn);
  float*    x2    = (float*)(ws + o_x2);
  uint16_t* ln2   = (uint16_t*)(ws + o_ln2);
  uint16_t* hact  = (uint16_t*)(ws + o_hact);

  float* out_tok  = (float*)d_out;
  float* out_idx  = out_tok + (size_t)M2 * C;
  float* out_mask = out_idx + (size_t)B * KP;

  // 1) weights -> bf16
  auto cvt = [&](const float* src, uint16_t* dst, size_t n) {
    cvt_f32_bf16<<<dim3((unsigned)((n + 255) / 256)), dim3(256), 0, stream>>>(src, dst, (int)n);
  };
  cvt(qkvw, Wqkv, (size_t)C * 3 * C);
  cvt(projw, Wproj, (size_t)C * C);
  cvt(fc1w, Wfc1, (size_t)C * 4 * C);
  cvt(fc2w, Wfc2, (size_t)4 * C * C);

  // 2) LN(norm1) over all tokens -> bf16
  ln768_kernel<<<dim3(M1), dim3(256), 0, stream>>>(tokens, n1w, n1b, normA);

  // 3) k_all = normA @ Wk (+bias) -> bf16
  gemm_wmma_bf16<<<dim3(C / 128, (M1 + 127) / 128), dim3(256), 0, stream>>>(
      normA, C, Wqkv + C, 3 * C, nullptr, kall, C, qkvb + C, nullptr, 0, M1, C, C, 3);
  // 4) q_cls = CLS rows @ Wq (+bias) -> f32   (CLS rows via lda = T*C)
  gemm_wmma_bf16<<<dim3(C / 128, 1), dim3(256), 0, stream>>>(
      normA, T * C, Wqkv, 3 * C, qcls, nullptr, C, qkvb, nullptr, 0, B, C, C, 0);

  // 5) CLS softmax scores, mean over heads
  cls_scores_kernel<<<dim3(B), dim3(640), 0, stream>>>(qcls, kall, scor);
  // 6) top-k (ascending) + index/mask outputs
  topk_kernel<<<dim3(B), dim3(576), 0, stream>>>(scor, pidx, kpos, out_idx, out_mask);
  // 7) gather kept tokens
  gather_kernel<<<dim3(M2), dim3(256), 0, stream>>>(tokens, kpos, x);

  // 8) LN(norm1) on kept tokens
  ln768_kernel<<<dim3(M2), dim3(256), 0, stream>>>(x, n1w, n1b, normx);
  // 9) qkv = normx @ Wqkv (+bias) -> bf16
  gemm_wmma_bf16<<<dim3(3 * C / 128, (M2 + 127) / 128), dim3(256), 0, stream>>>(
      normx, C, Wqkv, 3 * C, nullptr, qkvo, 3 * C, qkvb, nullptr, 0, M2, 3 * C, C, 3);
  // 10) attention per (b,h)
  attn_kernel<<<dim3(3, 12, B), dim3(128), 0, stream>>>(qkvo, attn);
  // 11) x2 = attn @ Wproj + proj_b + x
  gemm_wmma_bf16<<<dim3(C / 128, (M2 + 127) / 128), dim3(256), 0, stream>>>(
      attn, C, Wproj, C, x2, nullptr, C, projb, x, C, M2, C, C, 1);
  // 12) LN(norm2)
  ln768_kernel<<<dim3(M2), dim3(256), 0, stream>>>(x2, n2w, n2b, ln2);
  // 13) hact = gelu(ln2 @ Wfc1 + fc1_b) -> bf16
  gemm_wmma_bf16<<<dim3(4 * C / 128, (M2 + 127) / 128), dim3(256), 0, stream>>>(
      ln2, C, Wfc1, 4 * C, nullptr, hact, 4 * C, fc1b, nullptr, 0, M2, 4 * C, C, 2);
  // 14) out = hact @ Wfc2 + fc2_b + x2  -> d_out tokens
  gemm_wmma_bf16<<<dim3(C / 128, (M2 + 127) / 128), dim3(256), 0, stream>>>(
      hact, 4 * C, Wfc2, C, out_tok, nullptr, C, fc2b, x2, C, M2, C, 4 * C, 1);

  (void)in_sizes; (void)n_in; (void)out_size; (void)ws_size;
}
